// RNN_35012573397357
// MI455X (gfx1250) — matrix operand order repaired and checked by reading, not compiled, not run on hardware
//
#include <hip/hip_runtime.h>
#include <hip/hip_bf16.h>
#include <math.h>

typedef __attribute__((ext_vector_type(16))) __bf16 v16bf;
typedef __attribute__((ext_vector_type(8)))  float  v8f;

#define SEQ     16384
#define HID     1024
#define NWG     16
#define ROWS_WG (HID / NWG)        // 64 output rows per workgroup
#define MTILES  (ROWS_WG / 16)     // 4
#define KTILES  (HID / 32)         // 32
#define THREADS 256                // 8 wave32 waves

// Zero the inter-workgroup barrier state each launch (graph-replay safe).
__global__ void rnn_init(unsigned* bar) {
    if (threadIdx.x < 2) bar[threadIdx.x] = 0u;
}

__global__ __launch_bounds__(THREADS, 1)
void rnn_persistent(const float* __restrict__ X,
                    const float* __restrict__ Wm,
                    const float* __restrict__ bm,
                    const float* __restrict__ Wn,
                    const float* __restrict__ bn,
                    float* __restrict__ out,        // h_seq [SEQ*HID] then h_final [HID]
                    unsigned* __restrict__ bar,     // [0]=arrive count, [1]=generation
                    __bf16* __restrict__ hbuf)      // [2][HID] double-buffered h exchange
{
    // Wn slice pre-swizzled into exact 16x16x32 bf16 A-fragment order: 128 KB
    __shared__ __bf16 ldsA[MTILES * KTILES * 32 * 16];
    __shared__ __bf16 h_lds[HID];                   // current h (bf16)
    __shared__ float  red[MTILES][32][8];           // K-half partial C tiles
    __shared__ float  wm_s[ROWS_WG];
    __shared__ float  bmn_s[ROWS_WG];

    const int tid   = threadIdx.x;
    const int wg    = blockIdx.x;
    const int lane  = tid & 31;
    const int wave  = tid >> 5;
    const int mt    = wave & 3;    // M-tile owned by this wave
    const int khalf = wave >> 2;   // which half of K (0..511 / 512..1023)

    // ---- one-time: swizzle this WG's 64 rows of Wn into A-fragment layout ----
    // A (16x32 bf16) layout: lane L holds row M=L%16; lanes<16 carry K in
    // {0..7}u{16..23}, lanes>=16 carry K in {8..15}u{24..31}; element e=v*2+half
    // in the v16bf maps to Klocal = (v<4?0:16) + (L<16?0:8) + (v&3)*2 + half.
    for (int idx = tid; idx < MTILES * KTILES * 32 * 16; idx += THREADS) {
        int e    = idx & 15;
        int half = e & 1;
        int v    = e >> 1;
        int ln   = (idx >> 4) & 31;
        int kt   = (idx >> 9) & 31;
        int mtl  = (idx >> 14) & 3;
        int kl   = ((v < 4) ? 0 : 16) + ((ln < 16) ? 0 : 8) + ((v & 3) << 1) + half;
        int ml   = ln & 15;
        int row  = wg * ROWS_WG + mtl * 16 + ml;
        int col  = kt * 32 + kl;
        ldsA[idx] = (__bf16)Wn[row * HID + col];
    }
    for (int i = tid; i < HID; i += THREADS) h_lds[i] = (__bf16)0.0f;  // h0 = 0
    for (int i = tid; i < ROWS_WG; i += THREADS) {
        int j = wg * ROWS_WG + i;
        wm_s[i]  = Wm[j];            // IN_SIZE == 1
        bmn_s[i] = bm[j] + bn[j];
    }
    __syncthreads();

    for (int t = 0; t < SEQ; ++t) {
        // ---- C = Wn_slice * h over this wave's (M-tile, K-half) ----
        v8f c = {};
        #pragma unroll
        for (int k = 0; k < KTILES / 2; ++k) {
            int kt = khalf * (KTILES / 2) + k;
            const v16bf a =
                *reinterpret_cast<const v16bf*>(&ldsA[((mt * KTILES + kt) * 32 + lane) << 4]);
            // B (32x16 bf16): column N=lane%16, h[K] replicated across N; element i
            // carries K = (i<8?0:16) + (lane<16?0:8) + (i&7), matching A's K order.
            const int kb = kt * 32 + ((lane >> 4) << 3);
            v16bf b;
            #pragma unroll
            for (int i = 0; i < 16; ++i) b[i] = h_lds[kb + ((i >> 3) << 4) + (i & 7)];
            c = __builtin_amdgcn_wmma_f32_16x16x32_bf16(
                    false, a, false, b, (short)0, c, false, false);
        }
        __syncthreads();
        if (khalf == 1) {
            #pragma unroll
            for (int r = 0; r < 8; ++r) red[mt][lane][r] = c[r];
        }
        __syncthreads();
        // ---- combine K-halves, add pre, tanh, publish (columns are replicated:
        // lane 0 holds M=0..7 in c[0..7], lane 16 holds M=8..15) ----
        if (khalf == 0 && ((lane & 15) == 0)) {
            float xt = X[t * HID];          // x[t, 0, 0]
            int   hi = lane >> 4;
            #pragma unroll
            for (int r = 0; r < 8; ++r) {
                int   local = mt * 16 + hi * 8 + r;
                float y  = c[r] + red[mt][lane][r] + xt * wm_s[local] + bmn_s[local];
                float hn = tanhf(y);
                int   j  = wg * ROWS_WG + local;
                out[t * HID + j] = hn;                    // h_seq
                hbuf[(t & 1) * HID + j] = (__bf16)hn;     // exchange buffer
                if (t == SEQ - 1) out[SEQ * HID + j] = hn; // h_final
            }
        }
        // ---- inter-WGP sense-reversing barrier (release writes / acquire all) ----
        __syncthreads();
        if (tid == 0) {
            __threadfence();
            unsigned gen  = __hip_atomic_load(&bar[1], __ATOMIC_RELAXED,
                                              __HIP_MEMORY_SCOPE_AGENT);
            unsigned prev = __hip_atomic_fetch_add(&bar[0], 1u, __ATOMIC_ACQ_REL,
                                                   __HIP_MEMORY_SCOPE_AGENT);
            if (prev == NWG - 1) {
                __hip_atomic_store(&bar[0], 0u, __ATOMIC_RELAXED,
                                   __HIP_MEMORY_SCOPE_AGENT);
                __hip_atomic_fetch_add(&bar[1], 1u, __ATOMIC_RELEASE,
                                       __HIP_MEMORY_SCOPE_AGENT);
            } else {
                while (__hip_atomic_load(&bar[1], __ATOMIC_ACQUIRE,
                                         __HIP_MEMORY_SCOPE_AGENT) == gen)
                    __builtin_amdgcn_s_sleep(1);
            }
        }
        __syncthreads();
        // ---- pull everyone's h_{t+1} into LDS for the next step ----
        if (t + 1 < SEQ) {
            const __bf16* src = &hbuf[(t & 1) * HID];
            for (int i = tid; i < HID; i += THREADS) h_lds[i] = src[i];
            __syncthreads();
        }
    }
}

extern "C" void kernel_launch(void* const* d_in, const int* in_sizes, int n_in,
                              void* d_out, int out_size, void* d_ws, size_t ws_size,
                              hipStream_t stream) {
    const float* X  = (const float*)d_in[0];   // x  (16384*1024*1)
    const float* Wm = (const float*)d_in[1];   // Wm (1024*1)
    const float* bm = (const float*)d_in[2];   // bm (1024)
    const float* Wn = (const float*)d_in[3];   // Wn (1024*1024)
    const float* bn = (const float*)d_in[4];   // bn (1024)
    float*    out  = (float*)d_out;
    unsigned* bar  = (unsigned*)d_ws;
    __bf16*   hbuf = (__bf16*)((char*)d_ws + 16);

    rnn_init<<<1, 64, 0, stream>>>(bar);
    rnn_persistent<<<NWG, THREADS, 0, stream>>>(X, Wm, bm, Wn, bn, out, bar, hbuf);
}
